// MessageGraphAttention_76699525972271
// MI455X (gfx1250) — matrix operand order repaired and checked by reading, not compiled, not run on hardware
//
#include <hip/hip_runtime.h>

typedef __bf16 bf16_t;
typedef __attribute__((ext_vector_type(16))) __bf16 v16bf;
typedef __attribute__((ext_vector_type(8)))  float  v8f;

#define GN  1024
#define GF  64
#define GFE 32

// ---------------- workspace layout (bytes) ----------------
#define OFF_DEG   0                      // N int32
#define OFF_IDX   4096                   // N*N uint16   (2 MB)
#define OFF_C1    (OFF_IDX + GN*GN*2)    // N*F f32
#define OFF_C2    (OFF_C1 + GN*GF*4)     // N*F f32
#define OFF_BF    (OFF_C2 + GN*GF*4)     // 12288 bf16 (24 KB)
#define OFF_PP    (OFF_BF + 12288*2)     // N*F f32
#define OFF_Q     (OFF_PP + GN*GF*4)     // N*F f32

__device__ __forceinline__ float fast_sigmoid(float x) {
  // 1/(1+exp(-x)) with v_exp_f32 + v_rcp_f32 (avoid IEEE div expansion)
  return __builtin_amdgcn_rcpf(1.0f + __expf(-x));
}

// ---------------- kernel 1: per-row edge list from 0/1 adjacency ----------------
__global__ void k_edges(const float* __restrict__ A, int* __restrict__ deg,
                        unsigned short* __restrict__ idx) {
  int a = blockIdx.x;
  for (int b = threadIdx.x; b < GN; b += blockDim.x) {
    if (A[a * GN + b] != 0.0f) {
      int p = atomicAdd(&deg[a], 1);
      idx[a * GN + p] = (unsigned short)b;
    }
  }
}

// ---------------- kernel 2: c1 = H@W[0:64] + bias1, c2 = H@W_att[0:64] + bias2 ----------------
__global__ void k_proj(const float* __restrict__ H, const float* __restrict__ W,
                       const float* __restrict__ W_att,
                       const float* __restrict__ bias1, const float* __restrict__ bias2,
                       float* __restrict__ c1, float* __restrict__ c2) {
  int t = blockIdx.x * blockDim.x + threadIdx.x;
  if (t >= GN * GF) return;
  int a = t / GF, f = t % GF;
  float s1 = bias1[f], s2 = bias2[f];
  for (int k = 0; k < GF; ++k) {
    float h = H[a * GF + k];
    s1 += h * W[k * GF + f];
    s2 += h * W_att[k * GF + f];
  }
  c1[t] = s1;
  c2[t] = s2;
}

// ---------------- kernel 3: pack W2||W3 (K=96) into bf16 WMMA B fragments ----------------
// fragment index: ((mat*3 + s)*4 + ftile)*512 + lane*16 + slot
// B layout (32x16, 16-bit): lanes 0-15 hold K=0..7,16..23, lanes 16-31 hold K=8..15,24..31; col = lane&15
__global__ void k_pack(const float* __restrict__ W, const float* __restrict__ W_att,
                       bf16_t* __restrict__ bfrag) {
  int t = blockIdx.x * blockDim.x + threadIdx.x;
  if (t >= 12288) return;
  int slot = t & 15;
  int lane = (t >> 4) & 31;
  int ft   = (t >> 9) & 3;
  int s    = (t >> 11) % 3;
  int mat  = t / 6144;
  int klocal = ((lane < 16) ? 0 : 8) + (slot & 7) + ((slot & 8) ? 16 : 0);
  int krow   = 64 + s * 32 + klocal;        // rows 64..159 of W = [W2;W3]
  int col    = ft * 16 + (lane & 15);
  const float* src = mat ? W_att : W;
  bfrag[t] = (bf16_t)src[krow * GF + col];
}

// ---------------- main kernel ----------------
// 4 waves per node a (same block -> shared WGP$ for the gathered E/H rows);
// each wave owns ONE 16-feature tile. Per wave: 6 B fragments (48 VGPRs) in
// registers across the edge loop, 1 f32 accumulator, 3 bf16 A fragments.
__global__ void __launch_bounds__(256)
k_main(const float* __restrict__ H, const float* __restrict__ E,
       const int* __restrict__ deg, const unsigned short* __restrict__ idx,
       const float* __restrict__ c1, const float* __restrict__ c2,
       const float* __restrict__ bias1, const float* __restrict__ bias2,
       const bf16_t* __restrict__ bfrag_g, float* __restrict__ Ppool) {
  __shared__ __align__(32) bf16_t ldsB[12288];
  {
    const uint4* src = (const uint4*)bfrag_g;
    uint4* dst = (uint4*)ldsB;
    for (int i = threadIdx.x; i < 1536; i += blockDim.x) dst[i] = src[i];
  }
  __syncthreads();

  const int wave = threadIdx.x >> 5;              // 0..7
  const int lane = threadIdx.x & 31;
  const int a    = blockIdx.x * 2 + (wave >> 2);  // 2 nodes per block
  const int t    = wave & 3;                      // this wave's feature tile
  const int d    = deg[a];
  const int half = lane >> 4;
  const int l15  = lane & 15;
  const int koff = half * 8;

  // B fragments for this wave's feature tile, both matrices, 3 K-steps each.
  v16bf bm[3], ba[3];
  #pragma unroll
  for (int s = 0; s < 3; ++s) {
    bm[s] = *(const v16bf*)&ldsB[((0 * 3 + s) * 4 + t) * 512 + lane * 16];
    ba[s] = *(const v16bf*)&ldsB[((1 * 3 + s) * 4 + t) * 512 + lane * 16];
  }

  const float cmv = c1[a * GF + t * 16 + l15];
  const float cav = c2[a * GF + t * 16 + l15];

  const v8f zero8 = {0.f, 0.f, 0.f, 0.f, 0.f, 0.f, 0.f, 0.f};
  v8f acc = zero8;

  const int nch = (d + 15) >> 4;
  #pragma clang loop unroll(disable)
  for (int c = 0; c < nch; ++c) {
    int slot = c * 16 + l15;
    int sc   = slot < d ? slot : d - 1;
    int brow = (int)idx[a * GN + sc];

    const float* eptr = E + ((size_t)a * GN + (size_t)brow) * GFE;
    const float* hptr = H + (size_t)brow * GF;

    // A fragments (16-bit A 16x32 layout): lanes<16 hold K=koff..koff+7,16+koff..; lanes>=16 shifted by 8
    v16bf afE, afH0, afH1;
    #pragma unroll
    for (int i = 0; i < 8; ++i) {
      afE[i]    = (bf16_t)eptr[koff + i];
      afE[i+8]  = (bf16_t)eptr[16 + koff + i];
      afH0[i]   = (bf16_t)hptr[koff + i];
      afH0[i+8] = (bf16_t)hptr[16 + koff + i];
      afH1[i]   = (bf16_t)hptr[32 + koff + i];
      afH1[i+8] = (bf16_t)hptr[48 + koff + i];
    }

    v8f m = zero8;
    m = __builtin_amdgcn_wmma_f32_16x16x32_bf16(false, afH0, false, bm[0], (short)0, m, false, false);
    m = __builtin_amdgcn_wmma_f32_16x16x32_bf16(false, afH1, false, bm[1], (short)0, m, false, false);
    m = __builtin_amdgcn_wmma_f32_16x16x32_bf16(false, afE,  false, bm[2], (short)0, m, false, false);

    v8f ma = zero8;
    ma = __builtin_amdgcn_wmma_f32_16x16x32_bf16(false, afH0, false, ba[0], (short)0, ma, false, false);
    ma = __builtin_amdgcn_wmma_f32_16x16x32_bf16(false, afH1, false, ba[1], (short)0, ma, false, false);
    ma = __builtin_amdgcn_wmma_f32_16x16x32_bf16(false, afE,  false, ba[2], (short)0, ma, false, false);

    // gate + masked accumulate (rows of the tile = edge slots)
    #pragma unroll
    for (int r = 0; r < 8; ++r) {
      int sl   = c * 16 + half * 8 + r;
      float mm = m[r] + cmv;
      float aa = ma[r] + cav;
      float g  = fmaxf(mm, 0.f) * fast_sigmoid(aa);
      acc[r] += (sl < d) ? g : 0.f;
    }
  }

  // reduce rows (edges in this tile) + zero-edge correction term, then store
  {
    float s = acc[0] + acc[1] + acc[2] + acc[3] + acc[4] + acc[5] + acc[6] + acc[7];
    s += __shfl_xor(s, 16, 32);           // combine rows 0-7 (lanes<16) with rows 8-15
    float b1v = bias1[t * 16 + l15];
    float b2v = bias2[t * 16 + l15];
    float g0  = fmaxf(b1v, 0.f) * fast_sigmoid(b2v);
    s += (float)(GN - d) * g0;            // A==0 edges contribute constant G0
    if (lane < 16) Ppool[a * GF + t * 16 + l15] = s;
  }
}

// ---------------- small MLP layer: Y = relu(X@Wp + bias) ----------------
__global__ void k_mlp(const float* __restrict__ X, const float* __restrict__ Wp,
                      const float* __restrict__ bias, float* __restrict__ Y) {
  int t = blockIdx.x * blockDim.x + threadIdx.x;
  if (t >= GN * GF) return;
  int a = t / GF, f = t % GF;
  float s = bias[f];
  for (int k = 0; k < GF; ++k) s += X[a * GF + k] * Wp[k * GF + f];
  Y[t] = fmaxf(s, 0.f);
}

extern "C" void kernel_launch(void* const* d_in, const int* in_sizes, int n_in,
                              void* d_out, int out_size, void* d_ws, size_t ws_size,
                              hipStream_t stream) {
  const float* H     = (const float*)d_in[0];
  const float* A     = (const float*)d_in[1];
  const float* E     = (const float*)d_in[2];
  const float* W     = (const float*)d_in[3];
  const float* W_att = (const float*)d_in[4];
  const float* bias1 = (const float*)d_in[5];
  const float* bias2 = (const float*)d_in[6];
  const float* Wp1   = (const float*)d_in[7];
  const float* Wp2   = (const float*)d_in[8];
  const float* bias3 = (const float*)d_in[9];
  const float* bias4 = (const float*)d_in[10];

  char* ws = (char*)d_ws;
  int*            deg   = (int*)(ws + OFF_DEG);
  unsigned short* idx   = (unsigned short*)(ws + OFF_IDX);
  float*          c1    = (float*)(ws + OFF_C1);
  float*          c2    = (float*)(ws + OFF_C2);
  bf16_t*         bfrag = (bf16_t*)(ws + OFF_BF);
  float*          Ppool = (float*)(ws + OFF_PP);
  float*          Q     = (float*)(ws + OFF_Q);

  hipMemsetAsync(deg, 0, GN * sizeof(int), stream);
  k_edges<<<GN, 256, 0, stream>>>(A, deg, idx);
  k_proj<<<(GN * GF + 255) / 256, 256, 0, stream>>>(H, W, W_att, bias1, bias2, c1, c2);
  k_pack<<<(12288 + 255) / 256, 256, 0, stream>>>(W, W_att, bfrag);
  // 8 waves/block (wave32); 4 waves per node a -> 2 nodes/block -> grid = N/2
  k_main<<<GN / 2, 256, 0, stream>>>(H, E, deg, idx, c1, c2, bias1, bias2, bfrag, Ppool);
  k_mlp<<<(GN * GF + 255) / 256, 256, 0, stream>>>(Ppool, Wp1, bias3, Q);
  k_mlp<<<(GN * GF + 255) / 256, 256, 0, stream>>>(Q, Wp2, bias4, (float*)d_out);
}
